// CausalSelfAttention_33998961115861
// MI455X (gfx1250) — matrix operand order repaired and checked by reading, not compiled, not run on hardware
//
#include <hip/hip_runtime.h>
#include <hip/hip_bf16.h>
#include <stdint.h>

typedef __attribute__((ext_vector_type(16))) __bf16 v16bf;
typedef __attribute__((ext_vector_type(8)))  float  v8f;

#define WMMA_BF16(a, b, c) \
    __builtin_amdgcn_wmma_f32_16x16x32_bf16(false, (a), false, (b), (short)0, (c), false, false)

static __device__ __forceinline__ unsigned short f2bf(float f) {
    union { float f; unsigned u; } v; v.f = f;
    unsigned u = v.u;
    unsigned r = (u + 0x7FFFu + ((u >> 16) & 1u)) >> 16;  // round-to-nearest-even
    return (unsigned short)r;
}

// cross-half (xor-16) exchange via ds_swizzle: xor_mask=0x10, and_mask=0x1f
static __device__ __forceinline__ float xor16f(float x) {
    int i = __builtin_amdgcn_ds_swizzle(__builtin_bit_cast(int, x), 0x401f);
    return __builtin_bit_cast(float, i);
}

// ---------------------------------------------------------------------------
// f32 -> bf16 conversion kernel
// ---------------------------------------------------------------------------
__global__ void cvt_f32_bf16(const float* __restrict__ in,
                             unsigned short* __restrict__ out, int n) {
    int i = blockIdx.x * blockDim.x + threadIdx.x;
    if (i < n) out[i] = f2bf(in[i]);
}

// ---------------------------------------------------------------------------
// GEMM: C[M,N] = A[M,K](bf16) * B[K,N](bf16) + bias[N](f32)
// Columns < qScaleCols are additionally scaled by 0.125 (folds attention's
// 1/sqrt(DK) into the Q block of the QKV projection).
// 128 threads = 4 waves; block tile 128x64; wave tile 64x32 (4x2 WMMA tiles)
// -> 8 WMMAs per 12 fragment b128 loads per K-step, 64 accum VGPRs/lane.
// ---------------------------------------------------------------------------
template <bool BF16OUT>
__global__ void __launch_bounds__(128)
gemm_bf16(const unsigned short* __restrict__ A,
          const unsigned short* __restrict__ Bm,
          const float* __restrict__ bias,
          void* __restrict__ Out, int M, int N, int K, int qScaleCols) {
    __shared__ unsigned short Alds[128][32];  // [m][k]
    __shared__ unsigned short Blds[64][32];   // transposed: [n][k]

    const int tid  = threadIdx.x;
    const int lane = tid & 31;
    const int wave = tid >> 5;
    const int half = lane >> 4;
    const int n16  = lane & 15;

    const int rowBase = blockIdx.x * 128;
    const int colBase = blockIdx.y * 64;
    const int wRow = (wave >> 1) * 64;
    const int wCol = (wave & 1) * 32;

    v8f acc[4][2] = {};

    const int bkp = tid >> 3, bno = (tid & 7) * 8;  // B staging: k-pair, 8 n's

    // loop-carried staging pointers
    const unsigned short* aSrc = A + (size_t)(rowBase + tid) * K;          // row = tid
    const unsigned short* bSrc = Bm + (size_t)(2 * bkp) * N + colBase + bno;

    for (int kc = 0; kc < K; kc += 32) {
        __syncthreads();
        {   // stage A tile 128x32 (straight copy; thread = one row)
            const uint4* src = (const uint4*)(aSrc + kc);
            uint4 d0 = src[0], d1 = src[1];
            *(uint4*)&Alds[tid][0] = d0;
            *(uint4*)&Alds[tid][8] = d1;
        }
        {   // stage B tile 32x64 transposed -> [n][k], k-pairs packed as b32
            union { uint4 q; unsigned short u[8]; } b0, b1;
            b0.q = *(const uint4*)bSrc;
            b1.q = *(const uint4*)(bSrc + N);
#pragma unroll
            for (int i = 0; i < 8; ++i)
                *(unsigned*)&Blds[bno + i][2 * bkp] =
                    (unsigned)b0.u[i] | ((unsigned)b1.u[i] << 16);
        }
        if (kc + 32 < K) {
            __builtin_prefetch(aSrc + kc + 32, 0, 3);
            __builtin_prefetch(bSrc + (size_t)32 * N, 0, 3);
        }
        bSrc += (size_t)32 * N;
        __syncthreads();

        v16bf bfr[2];
#pragma unroll
        for (int ni = 0; ni < 2; ++ni) {   // B frag: col n=n16, K = half*16 + i
            const unsigned short* p = &Blds[wCol + ni * 16 + n16][0];
            uint4* d = (uint4*)&bfr[ni];
            d[0] = *(const uint4*)(p + half * 16);
            d[1] = *(const uint4*)(p + half * 16 + 8);
        }
#pragma unroll
        for (int mi = 0; mi < 4; ++mi) {   // A frag: row m=n16, K = half*16 + i
            v16bf afr;
            const unsigned short* p = &Alds[wRow + mi * 16 + n16][0];
            uint4* d = (uint4*)&afr;
            d[0] = *(const uint4*)(p + half * 16);
            d[1] = *(const uint4*)(p + half * 16 + 8);
#pragma unroll
            for (int ni = 0; ni < 2; ++ni)
                acc[mi][ni] = WMMA_BF16(afr, bfr[ni], acc[mi][ni]);
        }
    }

    // epilogue: + bias (optionally *0.125 for Q cols), store
#pragma unroll
    for (int mi = 0; mi < 4; ++mi)
#pragma unroll
        for (int ni = 0; ni < 2; ++ni) {
            const int col = colBase + wCol + ni * 16 + n16;
            const float bv = bias[col];
            const float sc = (col < qScaleCols) ? 0.125f : 1.0f;
#pragma unroll
            for (int r = 0; r < 8; ++r) {
                const int row = rowBase + wRow + mi * 16 + r + half * 8;
                const float v = (acc[mi][ni][r] + bv) * sc;
                if (BF16OUT)
                    ((unsigned short*)Out)[(size_t)row * N + col] = f2bf(v);
                else
                    ((float*)Out)[(size_t)row * N + col] = v;
            }
        }
}

// ---------------------------------------------------------------------------
// Flash attention, TRANSPOSED formulation:
//   S^T = K (16k x 64d) * Q^T (64d x 16q)   -> keys on VGPR axis, q on lanes
//   O^T = V^T (16d x 32k) * P^T (32k x 16q) -> d on VGPR axis, q on lanes
// Softmax reduction over keys = in-lane VALU + one xor-16 ds_swizzle.
// Block = (batch, head, 64 q rows); 4 waves, 16 q rows each.
// qkv: bf16 [B*S, 3H] (Q block pre-scaled by 1/8); out: bf16 [B*S, H]
// ---------------------------------------------------------------------------
__global__ void __launch_bounds__(128)
attn_kernel(const unsigned short* __restrict__ qkv,
            unsigned short* __restrict__ outb) {
    constexpr int S = 2048, H = 1024, NH = 16, DK = 64, TH3 = 3 * H;

    __shared__ unsigned short Klds[32][64];   // [key][d]
    __shared__ unsigned short Vlds[64][40];   // transposed [d][key], padded

    const int tid  = threadIdx.x;
    const int lane = tid & 31;
    const int wave = tid >> 5;
    const int half = lane >> 4;
    const int n16  = lane & 15;

    const int qtiles = S / 64;
    const int bid = blockIdx.x;
    const int qt  = bid % qtiles;
    const int h   = (bid / qtiles) % NH;
    const int bb  = bid / (qtiles * NH);
    const int qBase = qt * 64;
    const int qw    = qBase + wave * 16;     // this wave's first q row
    const size_t rowB = (size_t)bb * S;

    // Q^T B-fragments (2 d-chunks): b[i] = Q[qw+n16][c*32 + half*16 + i]
    v16bf qf[2];
    {
        const unsigned short* qp = qkv + (rowB + qw + n16) * TH3 + h * DK;
#pragma unroll
        for (int c = 0; c < 2; ++c) {
            uint4* d = (uint4*)&qf[c];
            d[0] = *(const uint4*)(qp + c * 32 + half * 16);
            d[1] = *(const uint4*)(qp + c * 32 + half * 16 + 8);
        }
    }

    v8f oacc[4] = {};
    float mstate = -3.0e38f, lstate = 0.0f;

    const int skey = tid >> 2, sdp = (tid & 3) * 16;   // K staging: copy
    const int vkp  = tid >> 3, vdp = (tid & 7) * 8;    // V staging: key-pair transpose

    // loop-carried staging pointers
    const unsigned short* kStage = qkv + (rowB + skey) * TH3 + H + h * DK + sdp;
    const unsigned short* vStage = qkv + (rowB + 2 * vkp) * TH3 + 2 * H + h * DK + vdp;
    const size_t chunkStep = (size_t)32 * TH3;

    const int nChunks = (qBase + 64) / 32;
    for (int ch = 0; ch < nChunks; ++ch) {
        const int kc = ch * 32;
        __syncthreads();
        {   // stage K (row-major copy)
            *(uint4*)&Klds[skey][sdp]     = *(const uint4*)kStage;
            *(uint4*)&Klds[skey][sdp + 8] = *(const uint4*)(kStage + 8);
        }
        {   // stage V transposed [d][key]: pack key-pairs into b32 stores
            union { uint4 q; unsigned short u[8]; } a0, a1;
            a0.q = *(const uint4*)vStage;
            a1.q = *(const uint4*)(vStage + TH3);
#pragma unroll
            for (int i = 0; i < 8; ++i)
                *(unsigned*)&Vlds[vdp + i][2 * vkp] =
                    (unsigned)a0.u[i] | ((unsigned)a1.u[i] << 16);
        }
        if (ch + 1 < nChunks) {
            __builtin_prefetch(kStage + chunkStep, 0, 3);
            __builtin_prefetch(vStage + chunkStep, 0, 3);
        }
        kStage += chunkStep;
        vStage += chunkStep;
        __syncthreads();

        // causal: this wave only needs chunks with kc <= qw+15 (wave-uniform)
        if (__builtin_amdgcn_readfirstlane((kc < qw + 16) ? 1 : 0)) {
            // S^T tiles: sub=0 -> keys [kc,kc+16), sub=1 -> [kc+16,kc+32)
            v8f st[2] = {};
#pragma unroll
            for (int sub = 0; sub < 2; ++sub)
#pragma unroll
                for (int c = 0; c < 2; ++c) {
                    v16bf kf;   // A frag: row m = key = sub*16+n16
                    const unsigned short* p = &Klds[sub * 16 + n16][c * 32];
                    uint4* d = (uint4*)&kf;
                    d[0] = *(const uint4*)(p + half * 16);
                    d[1] = *(const uint4*)(p + half * 16 + 8);
                    st[sub] = WMMA_BF16(kf, qf[c], st[sub]);
                }

            float sv[16];
#pragma unroll
            for (int r = 0; r < 8; ++r) { sv[r] = st[0][r]; sv[8 + r] = st[1][r]; }

            // mask only needed on diagonal chunks (wave-uniform test)
            if (!__builtin_amdgcn_readfirstlane((kc + 31 <= qw) ? 1 : 0)) {
                const int q = qw + n16;
#pragma unroll
                for (int r = 0; r < 8; ++r) {
                    if (kc + r + half * 8 > q)      sv[r]     = -3.0e38f;
                    if (kc + 16 + r + half * 8 > q) sv[8 + r] = -3.0e38f;
                }
            }

            // online softmax: in-lane reduce + one xor-16 exchange
            float mx = sv[0];
#pragma unroll
            for (int i = 1; i < 16; ++i) mx = fmaxf(mx, sv[i]);
            mx = fmaxf(mx, xor16f(mx));
            mx = fmaxf(mx, mstate);
            const float corr = __expf(mstate - mx);
            mstate = mx;

            float pv[16], sum = 0.0f;
#pragma unroll
            for (int i = 0; i < 16; ++i) { pv[i] = __expf(sv[i] - mx); sum += pv[i]; }
            sum += xor16f(sum);
            lstate = lstate * corr + sum;
#pragma unroll
            for (int t = 0; t < 4; ++t)
#pragma unroll
                for (int r = 0; r < 8; ++r) oacc[t][r] *= corr;

            // P^T B-fragment: keys half*16+i for column q=n16.
            // Exchange the sub the partner half needs via xor-16 swizzle.
            v16bf pf;
            unsigned short* pp = (unsigned short*)&pf;
#pragma unroll
            for (int r = 0; r < 8; ++r) {
                const float send = half ? pv[r] : pv[8 + r];
                const float ex   = xor16f(send);
                const float keep = half ? pv[8 + r] : pv[r];
                pp[r]     = f2bf(half ? ex : keep);
                pp[8 + r] = f2bf(half ? keep : ex);
            }

            // O^T += V^T (16d x 32k) * P^T per 16-wide d tile
#pragma unroll
            for (int t = 0; t < 4; ++t) {
                v16bf vf;   // A frag: row m = d = t*16+n16, K = keys half*16 + i
                const unsigned short* p = &Vlds[t * 16 + n16][0];
                uint4* d = (uint4*)&vf;
                d[0] = *(const uint4*)(p + half * 16);
                d[1] = *(const uint4*)(p + half * 16 + 8);
                oacc[t] = WMMA_BF16(vf, pf, oacc[t]);
            }
        }
    }

    // epilogue: per-lane scalar normalize; contiguous b128 stores
    const float inv = 1.0f / lstate;
    unsigned short* op = outb + (rowB + qw + n16) * H + h * DK;
#pragma unroll
    for (int t = 0; t < 4; ++t) {
        union { uint4 q; unsigned short u[8]; } ob;
#pragma unroll
        for (int r = 0; r < 8; ++r) ob.u[r] = f2bf(oacc[t][r] * inv);
        *(uint4*)(op + t * 16 + half * 8) = ob.q;
    }
}

// ---------------------------------------------------------------------------
extern "C" void kernel_launch(void* const* d_in, const int* in_sizes, int n_in,
                              void* d_out, int out_size, void* d_ws, size_t ws_size,
                              hipStream_t stream) {
    (void)in_sizes; (void)n_in; (void)out_size; (void)ws_size;
    const float* x      = (const float*)d_in[0];
    const float* w_attn = (const float*)d_in[1];
    const float* b_attn = (const float*)d_in[2];
    const float* w_proj = (const float*)d_in[3];
    const float* b_proj = (const float*)d_in[4];

    constexpr int Bb = 2, S = 2048, H = 1024;
    const int BS = Bb * S;  // 4096

    char* ws = (char*)d_ws;
    unsigned short* xb   = (unsigned short*)ws; ws += (size_t)BS * H * 2;
    unsigned short* wab  = (unsigned short*)ws; ws += (size_t)H * 3 * H * 2;
    unsigned short* wpb  = (unsigned short*)ws; ws += (size_t)H * H * 2;
    unsigned short* qkvb = (unsigned short*)ws; ws += (size_t)BS * 3 * H * 2;
    unsigned short* attb = (unsigned short*)ws;

    const int n1 = BS * H, n2 = H * 3 * H, n3 = H * H;
    cvt_f32_bf16<<<(n1 + 255) / 256, 256, 0, stream>>>(x, xb, n1);
    cvt_f32_bf16<<<(n2 + 255) / 256, 256, 0, stream>>>(w_attn, wab, n2);
    cvt_f32_bf16<<<(n3 + 255) / 256, 256, 0, stream>>>(w_proj, wpb, n3);

    dim3 g1(BS / 128, 3 * H / 64);
    gemm_bf16<true><<<g1, 128, 0, stream>>>(xb, wab, b_attn, qkvb, BS, 3 * H, H, H);

    attn_kernel<<<Bb * 16 * (S / 64), 128, 0, stream>>>(qkvb, attb);

    dim3 g2(BS / 128, H / 64);
    gemm_bf16<false><<<g2, 128, 0, stream>>>(attb, wpb, b_proj, d_out, BS, H, H, 0);
}
